// SummarizedConvolutionalSelfAttention_2723009266024
// MI455X (gfx1250) — matrix operand order repaired and checked by reading, compiled, not run on hardware
//
#include <hip/hip_runtime.h>
#include <hip/hip_bf16.h>

// ---------------------------------------------------------------------------
// SummarizedConvolutionalSelfAttention for MI455X (gfx1250, wave32, WMMA)
// All large matmuls + the grouped conv (implicit GEMM) on v_wmma_f32_16x16x32_f16.
// Each wave computes a 16x64 output tile (4 accumulators per A fragment);
// shared operand tiles are staged in LDS.
// ---------------------------------------------------------------------------

typedef __attribute__((ext_vector_type(8)))  _Float16 v8h_t;
typedef __attribute__((ext_vector_type(16))) _Float16 v16h_t;
typedef __attribute__((ext_vector_type(8)))  float    v8f_t;

#define BB     64
#define CC     256
#define HH     32
#define WWW    32
#define SS     1024      // H*W
#define OUTC   256
#define NHEAD  8
#define HD     32        // C / NHEAD
#define VD     64        // VALUE_DIM
#define NP     9         // K*K

#define LDS_LD 40        // LDS row stride in halves: 80B rows, 16B-aligned chunks

__device__ __forceinline__ v16h_t cat8(v8h_t a, v8h_t b) {
    v16h_t r;
#pragma unroll
    for (int i = 0; i < 8; ++i) { r[i] = a[i]; r[i + 8] = b[i]; }
    return r;
}

// A-matrix fragment: 16x32 f16. Lanes 0-15: row M=l, K {kb..kb+7, kb+16..kb+23};
// lanes 16-31: row M=l, K {kb+8..15, kb+24..31}.
__device__ __forceinline__ v16h_t load_a_frag(const _Float16* A, size_t lda,
                                              int m0, int kb, int lane) {
    int l = lane & 15, hi = lane >> 4;
    const _Float16* p = A + (size_t)(m0 + l) * lda + kb + hi * 8;
    v8h_t lo = *(const v8h_t*)(p);
    v8h_t hh = *(const v8h_t*)(p + 16);
    return cat8(lo, hh);
}

// B-matrix fragment from an LDS tile stored as [64 n][32 k] (row stride LDS_LD).
// Lanes 0-15: col N = nsub*16+l, K {0..7, 16..23}; lanes 16-31: K {8..15, 24..31}.
__device__ __forceinline__ v16h_t load_b_frag_lds(const _Float16* Bs, int nsub,
                                                  int lane) {
    int l = lane & 15, hi = lane >> 4;
    const _Float16* p = Bs + (size_t)(nsub * 16 + l) * LDS_LD + hi * 8;
    v8h_t lo = *(const v8h_t*)(p);
    v8h_t hh = *(const v8h_t*)(p + 16);
    return cat8(lo, hh);
}

// ----------------------------- small helpers -------------------------------

__global__ void f32_to_f16_kernel(const float* __restrict__ src,
                                  _Float16* __restrict__ dst, int n) {
    int i = blockIdx.x * blockDim.x + threadIdx.x;
    if (i < n) dst[i] = (_Float16)src[i];
}

// kv[b][s][c] = batch[b][c][s] + pe2d(s, c), converted to f16
__global__ void build_kv_kernel(const float* __restrict__ batch,
                                _Float16* __restrict__ kv_h) {
    size_t idx = (size_t)blockIdx.x * blockDim.x + threadIdx.x; // B*S*C
    int c = (int)(idx & 255);
    int s = (int)((idx >> 8) & 1023);
    int b = (int)(idx >> 18);
    int i = s >> 5, j = s & 31;
    float pe;
    if (c < 128) {
        int t = c >> 1;
        float f = __powf(10000.f, -(float)(2 * t) * (1.f / 128.f));
        float ang = (float)i * f;
        pe = (c & 1) ? __cosf(ang) : __sinf(ang);
    } else {
        int cc = c - 128;
        int t = cc >> 1;
        float f = __powf(10000.f, -(float)(2 * t) * (1.f / 128.f));
        float ang = (float)j * f;
        pe = (cc & 1) ? __cosf(ang) : __sinf(ang);
    }
    float x = batch[((size_t)b * CC + c) * SS + s];
    kv_h[idx] = (_Float16)(x + pe);
}

// baseplate (3,3,256) -> padded (16,256) f16, rows 9..15 zero
__global__ void build_qin_kernel(const float* __restrict__ baseplate,
                                 _Float16* __restrict__ qin_h) {
    int idx = blockIdx.x * blockDim.x + threadIdx.x; // 4096
    int row = idx >> 8, c = idx & 255;
    qin_h[idx] = (row < NP) ? (_Float16)baseplate[row * CC + c] : (_Float16)0.f;
}

// ------------------------- generic WMMA f16 GEMM ---------------------------
// out[m][n] = sum_k A[m][k] * W[n][k] + bias[n]
// grid.x = ceil(M/64), grid.y = N/64, block = 128.
// Block covers 64 M x 64 N: each wave owns a 16-row M sub-tile and computes a
// 16x64 strip (4 WMMA accs per A fragment). The 64x32 weight tile (shared by
// all waves) is staged once per K-step into LDS.
__global__ void __launch_bounds__(128)
gemm64_wmma_kernel(const _Float16* __restrict__ A, const _Float16* __restrict__ Wm,
                   const float* __restrict__ bias,
                   _Float16* __restrict__ outh, float* __restrict__ outf,
                   int Kd, int N, int M) {
    __shared__ _Float16 Bs[64 * LDS_LD];
    int lane = threadIdx.x & 31;
    int wave = threadIdx.x >> 5;
    int m0 = blockIdx.x * 64 + wave * 16;
    int nblk = blockIdx.y * 64;
    bool active = (m0 < M);

    v8f_t acc[4];
#pragma unroll
    for (int t = 0; t < 4; ++t) acc[t] = (v8f_t){0.f,0.f,0.f,0.f,0.f,0.f,0.f,0.f};

    for (int kb = 0; kb < Kd; kb += 32) {
        __syncthreads();
        {   // stage 64 rows x 32 k of W into LDS (each thread: one 32B chunk x2)
            int row = threadIdx.x >> 1, chunk = threadIdx.x & 1;
            const v8h_t* src =
                (const v8h_t*)(Wm + (size_t)(nblk + row) * Kd + kb + chunk * 16);
            v8h_t x0 = src[0], x1 = src[1];
            v8h_t* dst = (v8h_t*)(Bs + (size_t)row * LDS_LD + chunk * 16);
            dst[0] = x0; dst[1] = x1;
        }
        __syncthreads();
        if (active) {
            if (kb + 32 < Kd)
                __builtin_prefetch(A + (size_t)m0 * Kd + kb + 32, 0, 0);
            v16h_t a = load_a_frag(A, (size_t)Kd, m0, kb, lane);
#pragma unroll
            for (int t = 0; t < 4; ++t) {
                v16h_t b = load_b_frag_lds(Bs, t, lane);
                acc[t] = __builtin_amdgcn_wmma_f32_16x16x32_f16(
                    false, a, false, b, (short)0, acc[t], false, false);
            }
        }
    }

    if (active) {
        int l = lane & 15, hi = lane >> 4;
#pragma unroll
        for (int t = 0; t < 4; ++t) {
            int col = nblk + t * 16 + l;
            float bv = bias ? bias[col] : 0.f;
#pragma unroll
            for (int r = 0; r < 8; ++r) {
                int row = m0 + r + hi * 8;
                float v = acc[t][r] + bv;
                size_t o = (size_t)row * N + col;
                if (outh) outh[o] = (_Float16)v;
                if (outf) outf[o] = v;
            }
        }
    }
}

// ------------------------------ attention ----------------------------------
// grid (NHEAD, B), block 256. q_f: (16,256) f32 (rows 0..8 valid).
__global__ void __launch_bounds__(256)
attn_kernel(const float* __restrict__ qf, const _Float16* __restrict__ k_h,
            const _Float16* __restrict__ v_h, _Float16* __restrict__ ctx_h) {
    __shared__ float qs[NP * HD];
    __shared__ float sc[NP * SS];
    __shared__ float red[256];
    __shared__ float rowsum[NP];
    int t = threadIdx.x;
    int head = blockIdx.x, b = blockIdx.y;

    if (t < NP * HD) {
        int qi = t / HD, d = t % HD;
        qs[t] = qf[qi * CC + head * HD + d];
    }
    __syncthreads();

    for (int key = t; key < SS; key += 256) {
        const _Float16* kp = k_h + ((size_t)b * SS + key) * CC + head * HD;
        float kd[HD];
#pragma unroll
        for (int d = 0; d < HD; ++d) kd[d] = (float)kp[d];
#pragma unroll
        for (int qi = 0; qi < NP; ++qi) {
            float acc = 0.f;
#pragma unroll
            for (int d = 0; d < HD; ++d) acc += qs[qi * HD + d] * kd[d];
            sc[qi * SS + key] = acc * 0.17677669529663687f; // 1/sqrt(32)
        }
    }
    __syncthreads();

    for (int qi = 0; qi < NP; ++qi) {
        float m = -1e30f;
        for (int key = t; key < SS; key += 256) m = fmaxf(m, sc[qi * SS + key]);
        red[t] = m;
        __syncthreads();
        for (int s = 128; s > 0; s >>= 1) {
            if (t < s) red[t] = fmaxf(red[t], red[t + s]);
            __syncthreads();
        }
        m = red[0];
        __syncthreads();
        float ssum = 0.f;
        for (int key = t; key < SS; key += 256) {
            float e = __expf(sc[qi * SS + key] - m);
            sc[qi * SS + key] = e;
            ssum += e;
        }
        red[t] = ssum;
        __syncthreads();
        for (int s = 128; s > 0; s >>= 1) {
            if (t < s) red[t] += red[t + s];
            __syncthreads();
        }
        if (t == 0) rowsum[qi] = red[0];
        __syncthreads();
    }

    for (int o = t; o < NP * HD; o += 256) {
        int qi = o / HD, d = o % HD;
        const _Float16* vp = v_h + (size_t)b * SS * CC + head * HD + d;
        float acc = 0.f;
        for (int key = 0; key < SS; ++key)
            acc += sc[qi * SS + key] * (float)vp[(size_t)key * CC];
        ctx_h[((size_t)b * NP + qi) * CC + head * HD + d] =
            (_Float16)(acc / rowsum[qi]);
    }
}

// fbias[b][oc] = mean_p( reduced[b][p] . bg_w[oc] ) + bg_b[oc]
__global__ void fbias_kernel(const float* __restrict__ red_f,
                             const float* __restrict__ bg_w,
                             const float* __restrict__ bg_b,
                             float* __restrict__ fbias) {
    int idx = blockIdx.x * blockDim.x + threadIdx.x; // B*OUTC
    int b = idx >> 8, oc = idx & 255;
    const float* wp = bg_w + (size_t)oc * VD;
    float s = 0.f;
#pragma unroll
    for (int p = 0; p < NP; ++p) {
        const float* rp = red_f + ((size_t)b * NP + p) * VD;
        float d = 0.f;
#pragma unroll
        for (int v = 0; v < VD; ++v) d += rp[v] * wp[v];
        s += d;
    }
    fbias[idx] = s * (1.f / 9.f) + bg_b[oc];
}

// -------------------- per-sample conv as implicit WMMA GEMM ----------------
// out[b][oc][s] = sum_{p,c} fgout[b][p][oc*256+c] * X[b][c][y+dy][x+dx] + fbias
// grid (S/64, OUTC/64, B), block 128. Block covers 64 oc x 64 pixels:
// each wave owns a 16-oc sub-tile and computes 4 pixel sub-tiles per staged
// LDS activation tile (16 WMMA per staging pass).
__global__ void __launch_bounds__(128)
conv_wmma_kernel(const float* __restrict__ batch,
                 const _Float16* __restrict__ fgout,
                 const float* __restrict__ fbias,
                 float* __restrict__ out) {
    __shared__ _Float16 Xs[64 * LDS_LD];
    int b = blockIdx.z;
    int lane = threadIdx.x & 31;
    int wave = threadIdx.x >> 5;
    int oc0 = blockIdx.y * 64 + wave * 16;
    int s0 = blockIdx.x * 64;

    v8f_t acc[4];
#pragma unroll
    for (int t = 0; t < 4; ++t) acc[t] = (v8f_t){0.f,0.f,0.f,0.f,0.f,0.f,0.f,0.f};

    for (int p = 0; p < NP; ++p) {
        int dy = p / 3 - 1, dx = p % 3 - 1;
        const _Float16* Abase = fgout + ((size_t)(b * NP + p)) * (OUTC * CC);
        for (int cb = 0; cb < CC; cb += 32) {
            __syncthreads();
            // stage 64 pixels x 32 channels with zero padding, f32 -> f16
            for (int idx = threadIdx.x; idx < 64 * 32; idx += 128) {
                int c = idx >> 6, pixl = idx & 63;
                int s = s0 + pixl;
                int y = (s >> 5) + dy, x = (s & 31) + dx;
                float vv = 0.f;
                if ((unsigned)y < (unsigned)HH && (unsigned)x < (unsigned)WWW)
                    vv = batch[(((size_t)b * CC + cb + c) * HH + y) * WWW + x];
                Xs[pixl * LDS_LD + c] = (_Float16)vv;
            }
            __syncthreads();
            v16h_t a = load_a_frag(Abase, (size_t)CC, oc0, cb, lane);
#pragma unroll
            for (int t = 0; t < 4; ++t) {
                v16h_t bf = load_b_frag_lds(Xs, t, lane);
                acc[t] = __builtin_amdgcn_wmma_f32_16x16x32_f16(
                    false, a, false, bf, (short)0, acc[t], false, false);
            }
        }
    }

    int l = lane & 15, hi = lane >> 4;
#pragma unroll
    for (int t = 0; t < 4; ++t) {
        int col = s0 + t * 16 + l;
#pragma unroll
        for (int r8 = 0; r8 < 8; ++r8) {
            int oc = oc0 + r8 + hi * 8;
            out[((size_t)b * OUTC + oc) * SS + col] =
                acc[t][r8] + fbias[b * OUTC + oc];
        }
    }
}

// ------------------------------- launcher ----------------------------------

extern "C" void kernel_launch(void* const* d_in, const int* in_sizes, int n_in,
                              void* d_out, int out_size, void* d_ws, size_t ws_size,
                              hipStream_t stream) {
    const float* batch      = (const float*)d_in[0];
    const float* baseplate  = (const float*)d_in[1];
    const float* in_proj_w  = (const float*)d_in[2];
    const float* in_proj_b  = (const float*)d_in[3];
    const float* out_proj_w = (const float*)d_in[4];
    const float* out_proj_b = (const float*)d_in[5];
    const float* vr_w       = (const float*)d_in[6];
    const float* vr_b       = (const float*)d_in[7];
    const float* fg_w       = (const float*)d_in[8];
    const float* fg_b       = (const float*)d_in[9];
    const float* bg_w       = (const float*)d_in[10];
    const float* bg_b       = (const float*)d_in[11];
    float* out = (float*)d_out;

    char* ws = (char*)d_ws;
    // phase-1 region (kv/k/v), later aliased by fgout (75.5 MB < 100.7 MB)
    const size_t OFF_KV     = 0;                    // 65536*256 f16
    const size_t OFF_K      = 33554432;             // 65536*256 f16
    const size_t OFF_V      = 67108864;             // 65536*256 f16
    const size_t OFF_FGOUT  = 0;                    // 64*9*65536 f16 (aliases)
    const size_t OFF_INPROJ = 100663296;            // 768*256 f16
    const size_t OFF_OUTPRJ = 101056512;            // 256*256 f16
    const size_t OFF_VR     = 101187584;            // 64*256 f16
    const size_t OFF_FG     = 101220352;            // 65536*64 f16
    const size_t OFF_QIN    = 109608960;            // 16*256 f16
    const size_t OFF_QF     = 109617152;            // 16*256 f32
    const size_t OFF_CTX    = 109633536;            // 576*256 f16
    const size_t OFF_SUMM   = 109928448;            // 576*256 f16
    const size_t OFF_REDH   = 110223360;            // 576*64 f16
    const size_t OFF_REDF   = 110297088;            // 576*64 f32
    const size_t OFF_FBIAS  = 110444544;            // 64*256 f32

    _Float16* kv_h    = (_Float16*)(ws + OFF_KV);
    _Float16* k_h     = (_Float16*)(ws + OFF_K);
    _Float16* v_h     = (_Float16*)(ws + OFF_V);
    _Float16* fgout_h = (_Float16*)(ws + OFF_FGOUT);
    _Float16* inprj_h = (_Float16*)(ws + OFF_INPROJ);
    _Float16* outprj_h= (_Float16*)(ws + OFF_OUTPRJ);
    _Float16* vr_h    = (_Float16*)(ws + OFF_VR);
    _Float16* fg_h    = (_Float16*)(ws + OFF_FG);
    _Float16* qin_h   = (_Float16*)(ws + OFF_QIN);
    float*    q_f     = (float*)(ws + OFF_QF);
    _Float16* ctx_h   = (_Float16*)(ws + OFF_CTX);
    _Float16* summ_h  = (_Float16*)(ws + OFF_SUMM);
    _Float16* red_h   = (_Float16*)(ws + OFF_REDH);
    float*    red_f   = (float*)(ws + OFF_REDF);
    float*    fbias_f = (float*)(ws + OFF_FBIAS);

    // 1) weight conversions f32 -> f16
    f32_to_f16_kernel<<<(768 * 256 + 255) / 256, 256, 0, stream>>>(in_proj_w, inprj_h, 768 * 256);
    f32_to_f16_kernel<<<(256 * 256 + 255) / 256, 256, 0, stream>>>(out_proj_w, outprj_h, 256 * 256);
    f32_to_f16_kernel<<<(64 * 256 + 255) / 256, 256, 0, stream>>>(vr_w, vr_h, 64 * 256);
    f32_to_f16_kernel<<<(65536 * 64 + 255) / 256, 256, 0, stream>>>(fg_w, fg_h, 65536 * 64);

    // 2) kv = batch^T + positional encoding ; padded q input
    build_kv_kernel<<<(BB * SS * CC) / 256, 256, 0, stream>>>(batch, kv_h);
    build_qin_kernel<<<16, 256, 0, stream>>>(baseplate, qin_h);

    const _Float16* wq_h = inprj_h;
    const _Float16* wk_h = inprj_h + 256 * 256;
    const _Float16* wv_h = inprj_h + 512 * 256;

    // 3) projections (WMMA GEMMs)
    gemm64_wmma_kernel<<<dim3(1, 4), 128, 0, stream>>>(
        qin_h, wq_h, in_proj_b + 0, (_Float16*)nullptr, q_f, 256, 256, 16);
    gemm64_wmma_kernel<<<dim3(1024, 4), 128, 0, stream>>>(
        kv_h, wk_h, in_proj_b + 256, k_h, (float*)nullptr, 256, 256, 65536);
    gemm64_wmma_kernel<<<dim3(1024, 4), 128, 0, stream>>>(
        kv_h, wv_h, in_proj_b + 512, v_h, (float*)nullptr, 256, 256, 65536);

    // 4) attention -> ctx (B*9, 256)
    attn_kernel<<<dim3(NHEAD, BB), 256, 0, stream>>>(q_f, k_h, v_h, ctx_h);

    // 5) summaries = ctx @ out_proj^T + b   (576 x 256)
    gemm64_wmma_kernel<<<dim3(9, 4), 128, 0, stream>>>(
        ctx_h, outprj_h, out_proj_b, summ_h, (float*)nullptr, 256, 256, 576);

    // 6) reduced = summaries @ vr^T + b     (576 x 64), f16 + f32 copies
    gemm64_wmma_kernel<<<dim3(9, 1), 128, 0, stream>>>(
        summ_h, vr_h, vr_b, red_h, red_f, 256, 64, 576);

    // 7) fbias (uses f32 reduced)
    fbias_kernel<<<(BB * OUTC) / 256, 256, 0, stream>>>(red_f, bg_w, bg_b, fbias_f);

    // 8) filter generation: (576 x 65536) = reduced @ fg_w^T + fg_b
    //    (overwrites the kv/k/v region, which is dead by now)
    gemm64_wmma_kernel<<<dim3(9, 1024), 128, 0, stream>>>(
        red_h, fg_h, fg_b, fgout_h, (float*)nullptr, 64, 65536, 576);

    // 9) grouped conv with per-sample filters (implicit WMMA GEMM) + bias
    conv_wmma_kernel<<<dim3(SS / 64, OUTC / 64, BB), 128, 0, stream>>>(
        batch, fgout_h, fbias_f, out);
}
